// RelativePositionalAttention_5257039971101
// MI455X (gfx1250) — matrix operand order repaired and checked by reading, not compiled, not run on hardware
//
#include <hip/hip_runtime.h>
#include <hip/hip_bf16.h>
#include <stdint.h>

// ---------------------------------------------------------------------------
// MI455X (gfx1250) implementation notes:
//  * wave32 only; WMMA 16x16x32 bf16 -> f32 accumulate.
//  * Scores for one 16-row query tile live in LDS (16x1024 f32 = 64KB; WGP
//    has 320KB), so the B*N*S*S score tensor never touches HBM.
//  * rel_shift: pos_bias[i,j] = bd[i, S + j - i]; computed as two adjacent
//    16x16 bd WMMA tiles per (i-tile, j-tile), diagonal-gathered from LDS.
//  * V tiles for P*V are staged LDS-direct via GLOBAL_LOAD_ASYNC_TO_LDS_B128
//    (ASYNCcnt), with explicit s_wait_dscnt / s_wait_asynccnt fencing since
//    ASYNCcnt and DScnt are mutually unordered per the ISA.
// ---------------------------------------------------------------------------

typedef __bf16 v16bf __attribute__((ext_vector_type(16)));
typedef float  v8f   __attribute__((ext_vector_type(8)));

union FragBF { v16bf v; __bf16 e[16]; uint4 q[2]; };
union F8     { v8f   v; float  e[8]; };

static __device__ inline v8f wmma_bf16(v16bf a, v16bf b, v8f c) {
    // (neg_a, A, neg_b, B, c_mod, C, reuse_a, reuse_b)
    return __builtin_amdgcn_wmma_f32_16x16x32_bf16(false, a, false, b, (short)0, c, false, false);
}

static __device__ inline void load_contig16(FragBF& f, const __bf16* p) {
    f.q[0] = ((const uint4*)p)[0];
    f.q[1] = ((const uint4*)p)[1];
}

#define D_MODEL 1024
#define N_HEAD  16
#define D_HEAD  64
#define BSZ     4
#define SEQ     1024

// LDS layout offsets (bytes) for attn_kernel
#define SCORE_BYTES (16 * 1024 * 4)
#define BDB_BYTES   (16 * 32 * 4)
#define SBUF_BYTES  (32 * 4)
#define VT_OFF      (SCORE_BYTES + BDB_BYTES + SBUF_BYTES)   // 67712
#define SMEM_BYTES  (VT_OFF + 32 * 64 * 2)                   // 71808

// ---------------------------------------------------------------------------
// proj_gemm: C[M x 1024] = A[M x 1024] * W[1024 x 1024], bf16 epilogues.
// mode 0: q  -> qw/qr/qs = bf16(scale*(acc + rbias))   layout [b][n][i][h]
// mode 1/2: k/v -> bf16(acc + bias)                     layout [b][n][i][h]
// mode 3: pos -> bf16(acc)                              layout [n][t][h]
// One wave per 16x16 C tile; 4 waves per block.
// ---------------------------------------------------------------------------
__global__ void proj_gemm(const float* __restrict__ A, const float* __restrict__ W,
                          int M, int mode, float scale,
                          const float* __restrict__ b0, const float* __restrict__ b1,
                          const float* __restrict__ b2,
                          __bf16* __restrict__ o0, __bf16* __restrict__ o1,
                          __bf16* __restrict__ o2) {
    const int lane = threadIdx.x & 31;
    const int wave = threadIdx.x >> 5;
    const int tile = blockIdx.x * 4 + wave;
    const int i0 = (tile >> 6) * 16;          // 64 N-tiles (N = 1024)
    const int N0 = (tile & 63) * 16;
    if (i0 >= M) return;

    const int rowA  = lane & 15;
    const int khalf = (lane >> 4) * 16;
    const int colB  = lane & 15;
    const int kbase = (lane >> 4) * 16;

    const float* arow = A + (size_t)(i0 + rowA) * 1024;
    F8 acc = {};
    for (int kb = 0; kb < 1024; kb += 32) {
        __builtin_prefetch(arow + kb + 256, 0, 1);     // global_prefetch_b8
        FragBF a;
#pragma unroll
        for (int e = 0; e < 8; ++e) {
            a.e[e]     = (__bf16)arow[kb + khalf + e];
            a.e[8 + e] = (__bf16)arow[kb + khalf + 8 + e];
        }
        FragBF bf;
        const float* wp = W + (size_t)(kb + kbase) * 1024 + N0 + colB;
#pragma unroll
        for (int e = 0; e < 16; ++e) bf.e[e] = (__bf16)wp[(size_t)e * 1024];
        acc.v = wmma_bf16(a.v, bf.v, acc.v);
    }

#pragma unroll
    for (int v = 0; v < 8; ++v) {
        const int m   = i0 + (lane >> 4) * 8 + v;
        const int col = N0 + (lane & 15);
        const int n = col >> 6, h = col & 63;
        const float x = acc.e[v];
        if (mode == 3) {
            o0[((size_t)n * 2048 + m) * 64 + h] = (__bf16)x;
        } else {
            const size_t off = ((size_t)((m >> 10) * 16 + n) * 1024 + (m & 1023)) * 64 + h;
            if (mode == 0) {
                o0[off] = (__bf16)(scale * (x + b0[col]));
                o1[off] = (__bf16)(scale * (x + b1[col]));
                o2[off] = (__bf16)(scale * (x + b2[col]));
            } else {
                o0[off] = (__bf16)(x + b0[col]);
            }
        }
    }
}

// ---------------------------------------------------------------------------
// Fused attention: one wave (32 threads) per (b, n, 16 query rows).
// Phase 1: scores (content WMMA + rel-pos WMMA + seg + mask) -> LDS slab.
// Phase 2: softmax in LDS. Phase 3: P*V WMMA, V async-staged into LDS.
// ---------------------------------------------------------------------------
__global__ void attn_kernel(const __bf16* __restrict__ qw, const __bf16* __restrict__ qr,
                            const __bf16* __restrict__ qs, const __bf16* __restrict__ kh,
                            const __bf16* __restrict__ vh, const __bf16* __restrict__ rh,
                            const float* __restrict__ seg_embed,
                            const unsigned char* __restrict__ seg_mat,
                            const float* __restrict__ mask,
                            __bf16* __restrict__ av) {
    extern __shared__ char smem[];
    float* score = (float*)smem;                   // [16][1024]
    float* bdb   = score + 16 * 1024;              // [16][32]
    float* sbuf  = bdb + 16 * 32;                  // [16][2]
    __bf16* vt   = (__bf16*)(smem + VT_OFF);       // [32][64]

    const int blk = blockIdx.x;
    const int it = blk & 63, n = (blk >> 6) & 15, b = blk >> 10;
    const int i0 = it * 16;
    const int lane = threadIdx.x & 31;
    const int bn = b * 16 + n;

    // ---- segment-bias dot products (sb0 = diff, sb1 = same), 1 lane/row ----
    if (lane < 16) {
        const __bf16* qsr = qs + ((size_t)bn * 1024 + i0 + lane) * 64;
        float s0 = 0.f, s1 = 0.f;
        for (int h = 0; h < 64; ++h) {
            const float qv = (float)qsr[h];
            s0 += qv * seg_embed[(0 * 16 + n) * 64 + h];
            s1 += qv * seg_embed[(1 * 16 + n) * 64 + h];
        }
        sbuf[lane * 2 + 0] = s0;
        sbuf[lane * 2 + 1] = s1;
    }

    // ---- q fragments (reused across all 64 key tiles) ----
    const int rowA  = lane & 15;
    const int khalf = (lane >> 4) * 16;
    const int colB  = lane & 15;
    const int kbase = (lane >> 4) * 16;

    const __bf16* qwrow = qw + ((size_t)bn * 1024 + i0 + rowA) * 64;
    const __bf16* qrrow = qr + ((size_t)bn * 1024 + i0 + rowA) * 64;
    FragBF aw0, aw1, ar0, ar1;
    aw0.q[0] = ((const uint4*)(qwrow + khalf))[0];      aw0.q[1] = ((const uint4*)(qwrow + 8 + khalf))[0];
    aw1.q[0] = ((const uint4*)(qwrow + 32 + khalf))[0]; aw1.q[1] = ((const uint4*)(qwrow + 40 + khalf))[0];
    ar0.q[0] = ((const uint4*)(qrrow + khalf))[0];      ar0.q[1] = ((const uint4*)(qrrow + 8 + khalf))[0];
    ar1.q[0] = ((const uint4*)(qrrow + 32 + khalf))[0]; ar1.q[1] = ((const uint4*)(qrrow + 40 + khalf))[0];

    // ---- phase 1: score slab ----
    for (int jt = 0; jt < 64; ++jt) {
        const int j0 = jt * 16;
        // content = qw . K^T
        F8 cc = {};
        const __bf16* kp = kh + ((size_t)bn * 1024 + j0 + colB) * 64 + kbase;
        FragBF bk0, bk1;
        load_contig16(bk0, kp);
        load_contig16(bk1, kp + 32);
        cc.v = wmma_bf16(aw0.v, bk0.v, cc.v);
        cc.v = wmma_bf16(aw1.v, bk1.v, cc.v);

        // bd tiles: t0 = S + j0 - i0 - 16, band of 32 t values
        const int t0 = 1024 + j0 - i0 - 16;
#pragma unroll
        for (int tt = 0; tt < 2; ++tt) {
            F8 cb = {};
            const __bf16* rp = rh + ((size_t)n * 2048 + t0 + tt * 16 + colB) * 64 + kbase;
            FragBF br0, br1;
            load_contig16(br0, rp);
            load_contig16(br1, rp + 32);
            cb.v = wmma_bf16(ar0.v, br0.v, cb.v);
            cb.v = wmma_bf16(ar1.v, br1.v, cb.v);
            const int rc = (lane >> 4) * 8;
#pragma unroll
            for (int v = 0; v < 8; ++v)
                bdb[(rc + v) * 32 + tt * 16 + (lane & 15)] = cb.e[v];
        }

        // assemble: content + rel-shifted bd + seg bias - INF*mask
#pragma unroll
        for (int v = 0; v < 8; ++v) {
            const int r = (lane >> 4) * 8 + v;
            const int c = lane & 15;
            const int i = i0 + r, j = j0 + c;
            const size_t ij = ((size_t)b * 1024 + i) * 1024 + j;
            float sv = cc.e[v] + bdb[r * 32 + 16 + c - r]
                     + (seg_mat[ij] ? sbuf[r * 2 + 1] : sbuf[r * 2 + 0])
                     - 1.0e6f * mask[ij];
            score[r * 1024 + j] = sv;
        }
    }

    // ---- phase 2: softmax (2 lanes per row) ----
    {
        const int r = lane >> 1, half = lane & 1;
        float m = -3.4e38f;
        for (int c = half; c < 1024; c += 2) m = fmaxf(m, score[r * 1024 + c]);
        m = fmaxf(m, __shfl_xor(m, 1, 32));
        float s = 0.f;
        for (int c = half; c < 1024; c += 2) {
            const float ev = __expf(score[r * 1024 + c] - m);
            score[r * 1024 + c] = ev;
            s += ev;
        }
        s += __shfl_xor(s, 1, 32);
        const float inv = 1.0f / s;
        for (int c = half; c < 1024; c += 2) score[r * 1024 + c] *= inv;
    }

    // ---- phase 3: P * V (V tiles async-staged into LDS) ----
    F8 acc[4] = {{}, {}, {}, {}};
    for (int jt2 = 0; jt2 < 32; ++jt2) {
        const int jb = jt2 * 32;

        // Drain outstanding LDS reads of the previous tile before overwrite
        // (ASYNCcnt and DScnt are mutually unordered).
        asm volatile("s_wait_dscnt 0x0" ::: "memory");

        // Each lane moves one 64-element bf16 V row (128B) as 8x async b128.
        {
            const unsigned ldsoff =
                (unsigned)VT_OFF + (unsigned)lane * 128u;
            const unsigned long long ga =
                (unsigned long long)(uintptr_t)(vh + ((size_t)bn * 1024 + jb + lane) * 64);
#pragma unroll
            for (int rr = 0; rr < 8; ++rr) {
                asm volatile("global_load_async_to_lds_b128 %0, %1, off"
                             :: "v"(ldsoff + (unsigned)rr * 16u),
                                "v"(ga + (unsigned long long)rr * 16ull)
                             : "memory");
            }
        }
        asm volatile("s_wait_asynccnt 0x0" ::: "memory");

        // P fragment (f32 LDS -> bf16)
        FragBF pf;
#pragma unroll
        for (int e = 0; e < 8; ++e) {
            pf.e[e]     = (__bf16)score[rowA * 1024 + jb + khalf + e];
            pf.e[8 + e] = (__bf16)score[rowA * 1024 + jb + 8 + khalf + e];
        }
#pragma unroll
        for (int t = 0; t < 4; ++t) {
            FragBF bv;
            const int hc = t * 16 + (lane & 15);
#pragma unroll
            for (int e = 0; e < 16; ++e) bv.e[e] = vt[(kbase + e) * 64 + hc];
            acc[t].v = wmma_bf16(pf.v, bv.v, acc[t].v);
        }
    }

    // ---- write attn_vec, layout [b][i][n*64+h] ----
#pragma unroll
    for (int t = 0; t < 4; ++t)
#pragma unroll
        for (int v = 0; v < 8; ++v) {
            const int r = (lane >> 4) * 8 + v;
            const int c = lane & 15;
            av[((size_t)(b * 1024 + i0 + r)) * 1024 + n * 64 + t * 16 + c] = (__bf16)acc[t].e[v];
        }
}

// ---------------------------------------------------------------------------
// post_gemm: y[m,d] = q[m,d] + b_post[d] + sum_k av[m,k] * w_post[k,d]
// ---------------------------------------------------------------------------
__global__ void post_gemm(const __bf16* __restrict__ Av, const float* __restrict__ W,
                          const float* __restrict__ qin, const float* __restrict__ bpost,
                          float* __restrict__ y) {
    const int lane = threadIdx.x & 31;
    const int wave = threadIdx.x >> 5;
    const int tile = blockIdx.x * 4 + wave;
    const int i0 = (tile >> 6) * 16;
    const int N0 = (tile & 63) * 16;
    if (i0 >= 4096) return;

    const int rowA  = lane & 15;
    const int khalf = (lane >> 4) * 16;
    const int colB  = lane & 15;
    const int kbase = (lane >> 4) * 16;

    const __bf16* arow = Av + (size_t)(i0 + rowA) * 1024;
    F8 acc = {};
    for (int kb = 0; kb < 1024; kb += 32) {
        FragBF a;
        a.q[0] = ((const uint4*)(arow + kb + khalf))[0];
        a.q[1] = ((const uint4*)(arow + kb + 8 + khalf))[0];
        FragBF bf;
        const float* wp = W + (size_t)(kb + kbase) * 1024 + N0 + colB;
#pragma unroll
        for (int e = 0; e < 16; ++e) bf.e[e] = (__bf16)wp[(size_t)e * 1024];
        acc.v = wmma_bf16(a.v, bf.v, acc.v);
    }
#pragma unroll
    for (int v = 0; v < 8; ++v) {
        const int m   = i0 + (lane >> 4) * 8 + v;
        const int col = N0 + (lane & 15);
        y[(size_t)m * 1024 + col] = acc.e[v] + qin[(size_t)m * 1024 + col] + bpost[col];
    }
}

// ---------------------------------------------------------------------------
// LayerNorm over last dim (1024), one block (256 thr) per row.
// ---------------------------------------------------------------------------
__global__ void ln_kernel(const float* __restrict__ y, const float* __restrict__ g,
                          const float* __restrict__ bb, float* __restrict__ out) {
    __shared__ float red[256];
    const int row = blockIdx.x;
    const int tid = threadIdx.x;
    const float* yr = y + (size_t)row * 1024;

    float s = 0.f, s2 = 0.f;
#pragma unroll
    for (int j = 0; j < 4; ++j) {
        const float x = yr[tid + j * 256];
        s += x;
        s2 += x * x;
    }
    red[tid] = s; __syncthreads();
    for (int o = 128; o > 0; o >>= 1) { if (tid < o) red[tid] += red[tid + o]; __syncthreads(); }
    const float mu = red[0] * (1.0f / 1024.0f);
    __syncthreads();
    red[tid] = s2; __syncthreads();
    for (int o = 128; o > 0; o >>= 1) { if (tid < o) red[tid] += red[tid + o]; __syncthreads(); }
    const float var = red[0] * (1.0f / 1024.0f) - mu * mu;
    const float rstd = rsqrtf(var + 1e-5f);
#pragma unroll
    for (int j = 0; j < 4; ++j) {
        const int c = tid + j * 256;
        out[(size_t)row * 1024 + c] = (yr[c] - mu) * rstd * g[c] + bb[c];
    }
}

// ---------------------------------------------------------------------------
extern "C" void kernel_launch(void* const* d_in, const int* in_sizes, int n_in,
                              void* d_out, int out_size, void* d_ws, size_t ws_size,
                              hipStream_t stream) {
    (void)in_sizes; (void)n_in; (void)out_size; (void)ws_size;
    const float* q        = (const float*)d_in[0];
    const float* k        = (const float*)d_in[1];
    const float* v        = (const float*)d_in[2];
    const float* pos_enc  = (const float*)d_in[3];
    const unsigned char* seg_mat = (const unsigned char*)d_in[4];
    const float* attn_mask= (const float*)d_in[5];
    const float* wq       = (const float*)d_in[6];
    const float* wk       = (const float*)d_in[7];
    const float* bk       = (const float*)d_in[8];
    const float* wv       = (const float*)d_in[9];
    const float* bv       = (const float*)d_in[10];
    const float* r_w_bias = (const float*)d_in[11];
    const float* r_r_bias = (const float*)d_in[12];
    const float* r_kernel = (const float*)d_in[13];
    const float* r_s_bias = (const float*)d_in[14];
    const float* seg_embed= (const float*)d_in[15];
    const float* w_post   = (const float*)d_in[16];
    const float* b_post   = (const float*)d_in[17];
    const float* ln_g     = (const float*)d_in[18];
    const float* ln_b     = (const float*)d_in[19];

    char* ws = (char*)d_ws;
    __bf16* qw = (__bf16*)(ws + 0);                 // 8 MB each
    __bf16* qr = (__bf16*)(ws + 8388608);
    __bf16* qs = (__bf16*)(ws + 16777216);
    __bf16* kh = (__bf16*)(ws + 25165824);
    __bf16* vh = (__bf16*)(ws + 33554432);
    __bf16* rh = (__bf16*)(ws + 41943040);          // 4 MB
    __bf16* av = (__bf16*)(ws + 46137344);          // 8 MB
    float*  y  = (float*) (ws + 54525952);          // 16 MB

    const float scale = 0.125f;  // 1/sqrt(64)

    // projections (bf16 outputs, fused bias/scale)
    proj_gemm<<<4096, 128, 0, stream>>>(q, wq, 4096, 0, scale, r_w_bias, r_r_bias, r_s_bias, qw, qr, qs);
    proj_gemm<<<4096, 128, 0, stream>>>(k, wk, 4096, 1, 1.0f, bk, nullptr, nullptr, kh, nullptr, nullptr);
    proj_gemm<<<4096, 128, 0, stream>>>(v, wv, 4096, 2, 1.0f, bv, nullptr, nullptr, vh, nullptr, nullptr);
    proj_gemm<<<2048, 128, 0, stream>>>(pos_enc, r_kernel, 2048, 3, 1.0f, nullptr, nullptr, nullptr, rh, nullptr, nullptr);

    // fused attention: one wave per (b, n, 16 query rows); dynamic LDS
    attn_kernel<<<4096, 32, SMEM_BYTES, stream>>>(qw, qr, qs, kh, vh, rh, seg_embed, seg_mat, attn_mask, av);

    // post projection + residual, then layernorm
    post_gemm<<<4096, 128, 0, stream>>>(av, w_post, q, b_post, y);
    ln_kernel<<<4096, 256, 0, stream>>>(y, ln_g, ln_b, (float*)d_out);
}
